// SwinTransformerBlock_30245159699021
// MI455X (gfx1250) — compile-verified, matrix operand order checked by reading, not run on hardware
//
#include <hip/hip_runtime.h>

typedef __attribute__((ext_vector_type(16))) _Float16 v16h;
typedef __attribute__((ext_vector_type(8)))  _Float16 v8h;
typedef __attribute__((ext_vector_type(4)))  _Float16 v4h;
typedef __attribute__((ext_vector_type(8)))  float    v8f;

#define DIMC    128
#define NHEADS  4
#define HD      32
#define HH      56
#define WWID    56
#define NTOK    49          // tokens per window
#define HIDDEN  512
#define BATCH   64
#define TOK     (BATCH*HH*WWID)   // 200704
#define NWTOT   (BATCH*64)        // 4096 windows
#define QKSCALE 0.17677669529663687f  // 32^-0.5

// ---------------------------------------------------------------------------
// Fragment loader: 16x32 f16 tile from row-major memory (A operand layout;
// also B operand when the source is B-transposed / row-major K).
// lane m = lane&15 -> row; element e -> k = (e>=8?16:0) + 8*(lane>>4) + (e&7)
// => two contiguous 16-byte loads per lane.
// ---------------------------------------------------------------------------
__device__ __forceinline__ v16h load_frag16(const _Float16* tile, int ld, int lane) {
  const int m  = lane & 15;
  const int h8 = (lane >> 4) << 3;
  const _Float16* row = tile + (size_t)m * ld;
  v8h lo = *(const v8h*)(row + h8);
  v8h hi = *(const v8h*)(row + 16 + h8);
  v16h a;
#pragma unroll
  for (int e = 0; e < 8; ++e) { a[e] = lo[e]; a[e + 8] = hi[e]; }
  return a;
}

// ---------------------------------------------------------------------------
// Weight prep: W (K x N, f32) -> Wt (N x K, f16) so B-frag loads are b128.
// ---------------------------------------------------------------------------
__global__ void wprep_transpose_h(const float* __restrict__ W,
                                  _Float16* __restrict__ Wt, int K, int N) {
  int t = blockIdx.x * blockDim.x + threadIdx.x;
  if (t >= K * N) return;
  int n = t / K, k = t - n * K;
  Wt[t] = (_Float16)W[(size_t)k * N + n];
}

// ---------------------------------------------------------------------------
// LN1 + cyclic shift(-3,-3) + window partition, output f16 (windowed order).
// One wave per destination token; lane holds 4 of 128 channels.
// ---------------------------------------------------------------------------
__global__ __launch_bounds__(256) void ln1_shift_win(
    const float* __restrict__ x, const float* __restrict__ g,
    const float* __restrict__ bta, _Float16* __restrict__ xw) {
  const int token = blockIdx.x * 8 + (threadIdx.x >> 5);
  const int lane  = threadIdx.x & 31;
  const int win = token / NTOK;
  const int n   = token - win * NTOK;
  const int bI  = win >> 6, wid = win & 63;
  int gh = (wid >> 3) * 7 + n / 7 + 3; if (gh >= HH)   gh -= HH;
  int gw = (wid & 7)  * 7 + n % 7 + 3; if (gw >= WWID) gw -= WWID;
  const float4 v = ((const float4*)(x + ((size_t)bI * (HH*WWID) + gh * WWID + gw) * DIMC))[lane];
  float s = v.x + v.y + v.z + v.w;
#pragma unroll
  for (int t = 16; t >= 1; t >>= 1) s += __shfl_xor(s, t, 32);
  const float mean = s * (1.0f / 128.0f);
  const float dx = v.x - mean, dy = v.y - mean, dz = v.z - mean, dw = v.w - mean;
  float q = dx*dx + dy*dy + dz*dz + dw*dw;
#pragma unroll
  for (int t = 16; t >= 1; t >>= 1) q += __shfl_xor(q, t, 32);
  const float inv = rsqrtf(q * (1.0f / 128.0f) + 1e-5f);
  const float4 gv = ((const float4*)g)[lane];
  const float4 bv = ((const float4*)bta)[lane];
  v4h o;
  o[0] = (_Float16)(dx * inv * gv.x + bv.x);
  o[1] = (_Float16)(dy * inv * gv.y + bv.y);
  o[2] = (_Float16)(dz * inv * gv.z + bv.z);
  o[3] = (_Float16)(dw * inv * gv.w + bv.w);
  ((v4h*)(xw + (size_t)token * DIMC))[lane] = o;
}

// ---------------------------------------------------------------------------
// LN2 over residual x; also copies x into d_out (fc2 epilogue adds onto it).
// ---------------------------------------------------------------------------
__global__ __launch_bounds__(256) void ln2_copy(
    const float* __restrict__ xres, const float* __restrict__ g,
    const float* __restrict__ bta, _Float16* __restrict__ h16,
    float* __restrict__ dout) {
  const int token = blockIdx.x * 8 + (threadIdx.x >> 5);
  const int lane  = threadIdx.x & 31;
  const float4 v = ((const float4*)(xres + (size_t)token * DIMC))[lane];
  ((float4*)(dout + (size_t)token * DIMC))[lane] = v;
  float s = v.x + v.y + v.z + v.w;
#pragma unroll
  for (int t = 16; t >= 1; t >>= 1) s += __shfl_xor(s, t, 32);
  const float mean = s * (1.0f / 128.0f);
  const float dx = v.x - mean, dy = v.y - mean, dz = v.z - mean, dw = v.w - mean;
  float q = dx*dx + dy*dy + dz*dz + dw*dw;
#pragma unroll
  for (int t = 16; t >= 1; t >>= 1) q += __shfl_xor(q, t, 32);
  const float inv = rsqrtf(q * (1.0f / 128.0f) + 1e-5f);
  const float4 gv = ((const float4*)g)[lane];
  const float4 bv = ((const float4*)bta)[lane];
  v4h o;
  o[0] = (_Float16)(dx * inv * gv.x + bv.x);
  o[1] = (_Float16)(dy * inv * gv.y + bv.y);
  o[2] = (_Float16)(dz * inv * gv.z + bv.z);
  o[3] = (_Float16)(dw * inv * gv.w + bv.w);
  ((v4h*)(h16 + (size_t)token * DIMC))[lane] = o;
}

// ---------------------------------------------------------------------------
// Generic WMMA GEMM: C[M,N] = A[M,K] * Bt[N,K]^T + bias, wave = 16x128 tile
// (one A-frag feeds 8 WMMAs; grid.y = N/128 so FC2/proj are single-pass in A).
// EPI: 0 = store f16, 1 = GELU(exact) then f16,
//      2 = window-reverse + unshift scatter, += residual, store f32,
//      3 = accumulate f32 into out (+=).
// ---------------------------------------------------------------------------
template <int EPI>
__global__ __launch_bounds__(256) void gemm16x128(
    const _Float16* __restrict__ A, const _Float16* __restrict__ Bt,
    const float* __restrict__ bias, void* __restrict__ out,
    const float* __restrict__ resid_in, float* __restrict__ resid_out,
    int M, int N, int K) {
  const int lane = threadIdx.x & 31;
  const int wave = threadIdx.x >> 5;
  const int row0 = (blockIdx.x * 8 + wave) * 16;
  const int col0 = blockIdx.y * 128;
  if (row0 >= M) return;                       // wave-uniform (EXEC stays full)
  v8f acc[8] = {};
  for (int k0 = 0; k0 < K; k0 += 32) {
    v16h a = load_frag16(A + (size_t)row0 * K + k0, K, lane);
#pragma unroll
    for (int j = 0; j < 8; ++j) {
      v16h b = load_frag16(Bt + (size_t)(col0 + j * 16) * K + k0, K, lane);
      acc[j] = __builtin_amdgcn_wmma_f32_16x16x32_f16(
          false, a, false, b, (short)0, acc[j], false, false);
    }
  }
  const int m16 = lane & 15;
  const int h8  = (lane >> 4) << 3;
#pragma unroll
  for (int j = 0; j < 8; ++j) {
    const int col = col0 + j * 16 + m16;
    const float bv = bias[col];
#pragma unroll
    for (int r = 0; r < 8; ++r) {
      const int row = row0 + r + h8;
      float v = acc[j][r] + bv;
      if (EPI == 0) {
        ((_Float16*)out)[(size_t)row * N + col] = (_Float16)v;
      } else if (EPI == 1) {
        v = 0.5f * v * (1.0f + erff(v * 0.70710678118654752f));
        ((_Float16*)out)[(size_t)row * N + col] = (_Float16)v;
      } else if (EPI == 2) {
        const int win = row / NTOK;
        const int n   = row - win * NTOK;
        const int bI  = win >> 6, wid = win & 63;
        int gh = (wid >> 3) * 7 + n / 7 + 3; if (gh >= HH)   gh -= HH;
        int gw = (wid & 7)  * 7 + n % 7 + 3; if (gw >= WWID) gw -= WWID;
        const size_t orig = (size_t)bI * (HH*WWID) + gh * WWID + gw;
        resid_out[orig * DIMC + col] = resid_in[orig * DIMC + col] + v;
      } else {
        ((float*)out)[(size_t)row * N + col] += v;
      }
    }
  }
}

// ---------------------------------------------------------------------------
// Windowed attention: one wave per (window, head); 4 waves per block.
// S = scale*Q*K^T + rel-pos-bias + shift mask; softmax; O = P*V.
// N=49 padded to 64 (masked). P staged in LDS (f16 64x64 per wave).
// ---------------------------------------------------------------------------
__global__ __launch_bounds__(128) void swin_attn(
    const _Float16* __restrict__ qkv, const float* __restrict__ rpb,
    _Float16* __restrict__ outh) {
  __shared__ _Float16 Pl[NHEADS][64 * 64];
  const int head = threadIdx.x >> 5;
  const int lane = threadIdx.x & 31;
  const int win  = blockIdx.x;
  const int wi   = win & 63;
  const int wh7  = (wi >> 3) * 7;
  const int ww7  = (wi & 7)  * 7;
  const int m    = lane & 15;
  const int h8   = (lane >> 4) << 3;
  const _Float16* base = qkv + (size_t)win * NTOK * 384;

  // -------- stage 1: S tiles (4x4 of 16x16), one K-step of 32 each --------
  v16h kf[4];
#pragma unroll
  for (int t = 0; t < 4; ++t) {
    const int kr = t * 16 + m;                 // key index (lane column)
    v16h f = {};
    if (kr < NTOK) {
      const _Float16* rp = base + (size_t)kr * 384 + 128 + head * HD;
      v8h lo = *(const v8h*)(rp + h8);
      v8h hi = *(const v8h*)(rp + 16 + h8);
#pragma unroll
      for (int e = 0; e < 8; ++e) { f[e] = lo[e]; f[e + 8] = hi[e]; }
    }
    kf[t] = f;
  }
  v8f S[4][4];
#pragma unroll
  for (int mi = 0; mi < 4; ++mi) {
    const int qr = mi * 16 + m;                // query index (lane row)
    v16h qf = {};
    if (qr < NTOK) {
      const _Float16* rp = base + (size_t)qr * 384 + head * HD;
      v8h lo = *(const v8h*)(rp + h8);
      v8h hi = *(const v8h*)(rp + 16 + h8);
#pragma unroll
      for (int e = 0; e < 8; ++e) { qf[e] = lo[e]; qf[e + 8] = hi[e]; }
    }
#pragma unroll
    for (int ni = 0; ni < 4; ++ni) {
      v8f z = {};
      S[mi][ni] = __builtin_amdgcn_wmma_f32_16x16x32_f16(
          false, qf, false, kf[ni], (short)0, z, false, false);
    }
  }

  // -------- bias + mask + masked softmax; P -> LDS f16 --------
#pragma unroll
  for (int mi = 0; mi < 4; ++mi) {
#pragma unroll
    for (int r = 0; r < 8; ++r) {
      const int rq  = mi * 16 + r + h8;        // query token (row of C tile)
      const int qdr = rq / 7, qdc = rq % 7;
      const int qh = wh7 + qdr, qw = ww7 + qdc;
      const int regq = (qh < 49 ? 0 : (qh < 53 ? 1 : 2)) * 3 +
                       (qw < 49 ? 0 : (qw < 53 ? 1 : 2));
      float vals[4];
#pragma unroll
      for (int ni = 0; ni < 4; ++ni) {
        const int ck = ni * 16 + m;            // key token (col of C tile)
        float v;
        if (ck < NTOK && rq < NTOK) {
          const int kdr = ck / 7, kdc = ck % 7;
          v = S[mi][ni][r] * QKSCALE;
          v += rpb[((qdr - kdr + 6) * 13 + (qdc - kdc + 6)) * NHEADS + head];
          const int kh = wh7 + kdr, kw = ww7 + kdc;
          const int regk = (kh < 49 ? 0 : (kh < 53 ? 1 : 2)) * 3 +
                           (kw < 49 ? 0 : (kw < 53 ? 1 : 2));
          if (regq != regk) v -= 100.0f;
        } else {
          v = (ck < NTOK) ? 0.0f : -1e30f;     // pad cols -> -inf; pad rows harmless
        }
        vals[ni] = v;
      }
      float mx = fmaxf(fmaxf(vals[0], vals[1]), fmaxf(vals[2], vals[3]));
#pragma unroll
      for (int s = 1; s < 16; s <<= 1) mx = fmaxf(mx, __shfl_xor(mx, s, 32));
      float e0[4], sum = 0.0f;
#pragma unroll
      for (int ni = 0; ni < 4; ++ni) {
        e0[ni] = ((ni * 16 + m) < NTOK) ? __expf(vals[ni] - mx) : 0.0f;
        sum += e0[ni];
      }
#pragma unroll
      for (int s = 1; s < 16; s <<= 1) sum += __shfl_xor(sum, s, 32);
      const float inv = 1.0f / sum;
#pragma unroll
      for (int ni = 0; ni < 4; ++ni)
        Pl[head][rq * 64 + ni * 16 + m] = (_Float16)(e0[ni] * inv);
    }
  }
  __syncthreads();

  // -------- stage 2: O = P * V  (K padded to 64; P pad cols are zero) -----
  v8f O[4][2] = {};
  const _Float16* P = &Pl[head][0];
#pragma unroll
  for (int ki = 0; ki < 2; ++ki) {
    v16h vf[2];
#pragma unroll
    for (int nj = 0; nj < 2; ++nj) {
      v16h f = {};
#pragma unroll
      for (int e = 0; e < 16; ++e) {
        const int k = ki * 32 + (e >> 3) * 16 + h8 + (e & 7);
        if (k < NTOK)
          f[e] = base[(size_t)k * 384 + 256 + head * HD + nj * 16 + m];
      }
      vf[nj] = f;
    }
#pragma unroll
    for (int mi = 0; mi < 4; ++mi) {
      v16h pf = load_frag16(P + mi * 16 * 64 + ki * 32, 64, lane);
#pragma unroll
      for (int nj = 0; nj < 2; ++nj)
        O[mi][nj] = __builtin_amdgcn_wmma_f32_16x16x32_f16(
            false, pf, false, vf[nj], (short)0, O[mi][nj], false, false);
    }
  }
#pragma unroll
  for (int mi = 0; mi < 4; ++mi) {
#pragma unroll
    for (int r = 0; r < 8; ++r) {
      const int row = mi * 16 + r + h8;
      if (row < NTOK) {
        const size_t tok = (size_t)win * NTOK + row;
#pragma unroll
        for (int nj = 0; nj < 2; ++nj)
          outh[tok * DIMC + head * HD + nj * 16 + m] = (_Float16)O[mi][nj][r];
      }
    }
  }
}

// ---------------------------------------------------------------------------
extern "C" void kernel_launch(void* const* d_in, const int* in_sizes, int n_in,
                              void* d_out, int out_size, void* d_ws, size_t ws_size,
                              hipStream_t stream) {
  (void)in_sizes; (void)n_in; (void)out_size; (void)ws_size;
  const float* x      = (const float*)d_in[0];
  const float* n1g    = (const float*)d_in[1];
  const float* n1b    = (const float*)d_in[2];
  const float* qkv_w  = (const float*)d_in[3];
  const float* qkv_b  = (const float*)d_in[4];
  const float* rpb    = (const float*)d_in[5];
  const float* proj_w = (const float*)d_in[6];
  const float* proj_b = (const float*)d_in[7];
  const float* n2g    = (const float*)d_in[8];
  const float* n2b    = (const float*)d_in[9];
  const float* fc1_w  = (const float*)d_in[10];
  const float* fc1_b  = (const float*)d_in[11];
  const float* fc2_w  = (const float*)d_in[12];
  const float* fc2_b  = (const float*)d_in[13];
  float* out = (float*)d_out;

  // Workspace carving (aliased buffers, ordering enforced by stream order):
  //   bufA: xw(f16) -> attn_out(f16) -> ln2(f16)       [TOK*128 f16]
  //   bufB: qkv(f16, TOK*384) -> fc1 act(f16, TOK*512) [TOK*512 f16]
  //   xres: x + attn residual (f32)                    [TOK*128 f32]
  char* ws = (char*)d_ws;
  size_t off = 0;
  auto carve = [&](size_t bytes) {
    void* p = ws + off;
    off = (off + bytes + 255) & ~(size_t)255;
    return p;
  };
  _Float16* qkv_wt  = (_Float16*)carve((size_t)384 * 128 * 2);
  _Float16* proj_wt = (_Float16*)carve((size_t)128 * 128 * 2);
  _Float16* fc1_wt  = (_Float16*)carve((size_t)512 * 128 * 2);
  _Float16* fc2_wt  = (_Float16*)carve((size_t)128 * 512 * 2);
  _Float16* bufA    = (_Float16*)carve((size_t)TOK * 128 * 2);
  _Float16* bufB    = (_Float16*)carve((size_t)TOK * 512 * 2);
  float*    xres    = (float*)   carve((size_t)TOK * 128 * 4);

  // 1) weight prep (f32 -> transposed f16)
  wprep_transpose_h<<<(384*128 + 255) / 256, 256, 0, stream>>>(qkv_w,  qkv_wt,  128, 384);
  wprep_transpose_h<<<(128*128 + 255) / 256, 256, 0, stream>>>(proj_w, proj_wt, 128, 128);
  wprep_transpose_h<<<(128*512 + 255) / 256, 256, 0, stream>>>(fc1_w,  fc1_wt,  128, 512);
  wprep_transpose_h<<<(512*128 + 255) / 256, 256, 0, stream>>>(fc2_w,  fc2_wt,  512, 128);

  // 2) LN1 + shift + window partition -> bufA (f16, windowed token order)
  ln1_shift_win<<<TOK / 8, 256, 0, stream>>>(x, n1g, n1b, bufA);

  // 3) QKV GEMM: (TOK x 128) * (128 x 384) -> bufB f16
  gemm16x128<0><<<dim3(TOK / 128, 384 / 128), 256, 0, stream>>>(
      bufA, qkv_wt, qkv_b, bufB, nullptr, nullptr, TOK, 384, 128);

  // 4) windowed attention -> bufA f16 (heads concatenated)
  swin_attn<<<NWTOT, 128, 0, stream>>>(bufB, rpb, bufA);

  // 5) proj GEMM + window reverse + unshift + residual -> xres f32 (1 pass)
  gemm16x128<2><<<dim3(TOK / 128, 128 / 128), 256, 0, stream>>>(
      bufA, proj_wt, proj_b, nullptr, x, xres, TOK, 128, 128);

  // 6) LN2 -> bufA f16; copy xres into d_out
  ln2_copy<<<TOK / 8, 256, 0, stream>>>(xres, n2g, n2b, bufA, out);

  // 7) FC1 + exact GELU -> bufB f16
  gemm16x128<1><<<dim3(TOK / 128, 512 / 128), 256, 0, stream>>>(
      bufA, fc1_wt, fc1_b, bufB, nullptr, nullptr, TOK, 512, 128);

  // 8) FC2, accumulate into d_out (single pass over the 205 MB activation)
  gemm16x128<3><<<dim3(TOK / 128, 128 / 128), 256, 0, stream>>>(
      bufB, fc2_wt, fc2_b, out, nullptr, nullptr, TOK, 128, 512);
}
